// DecoderLayer_39144331936379
// MI455X (gfx1250) — matrix operand order repaired and checked by reading, not compiled, not run on hardware
//
#include <hip/hip_runtime.h>
#include <hip/hip_bf16.h>

typedef __bf16 bf16_t;
typedef __attribute__((ext_vector_type(16))) __bf16 v16bf;
typedef __attribute__((ext_vector_type(8)))  __bf16 v8bf;
typedef __attribute__((ext_vector_type(8)))  float   v8f;

// ---------------- wave reductions (wave32) ----------------
__device__ __forceinline__ float waveMax(float v) {
#pragma unroll
  for (int m = 16; m >= 1; m >>= 1) v = fmaxf(v, __shfl_xor(v, m, 32));
  return v;
}
__device__ __forceinline__ float waveSum(float v) {
#pragma unroll
  for (int m = 16; m >= 1; m >>= 1) v += __shfl_xor(v, m, 32);
  return v;
}

// CDNA5 async global->LDS copy (ASYNCcnt path). lds_addr is the wave-relative
// LDS byte address = low 32 bits of the flat shared-pointer (ISA 10.2 flat->LDS).
__device__ __forceinline__ void async_copy16(unsigned lds_addr, const void* gptr) {
  asm volatile("global_load_async_to_lds_b128 %0, %1, off"
               :: "v"(lds_addr), "v"(gptr)
               : "memory");
}

// ---------------- bf16 WMMA GEMM ----------------
// C[M,N] = A[M,K] * B[K,N] + bias ; A row-major bf16, Bt = B^T row-major bf16 ([N][K]).
// Block: 128 threads (4 waves). Block tile 128x128, k-step 32, double-buffered LDS
// filled with global_load_async_to_lds_b128. Wave tile 64x64 -> 4x4 accumulators
// of 16x16 (v_wmma_f32_16x16x32_bf16): 16 WMMA per k-step per wave.
#define TM 128
#define TN 128
#define TK 32
#define LDK 40   // TK + 8 bf16 pad (80B row stride, 16B aligned)

__global__ __launch_bounds__(128) void gemm_bf16_wmma(
    const bf16_t* __restrict__ A, const bf16_t* __restrict__ Bt,
    const float* __restrict__ bias, float* __restrict__ C,
    bf16_t* __restrict__ Cb, int Mdim, int Ndim, int Kdim, int relu)
{
  __shared__ __align__(16) bf16_t As[2][TM * LDK];
  __shared__ __align__(16) bf16_t Bs[2][TN * LDK];

  const int tid  = threadIdx.x;
  const int lane = tid & 31;
  const int wave = tid >> 5;
  const int m0 = blockIdx.y * TM;
  const int n0 = blockIdx.x * TN;
  const int wm = (wave & 1) * 64;   // wave row offset in block tile
  const int wn = (wave >> 1) * 64;  // wave col offset in block tile
  const int lrow = lane & 15;
  const int kA = (lane < 16) ? 0 : 8;    // A frag: K 0..7/16..23 vs 8..15/24..31
  const int kB = (lane < 16) ? 0 : 8;    // B frag: K 0..7/16..23 vs 8..15/24..31

  v8f acc[4][4];
#pragma unroll
  for (int i = 0; i < 4; ++i)
#pragma unroll
    for (int j = 0; j < 4; ++j)
#pragma unroll
      for (int e = 0; e < 8; ++e) acc[i][j][e] = 0.0f;

  // issue one 128x32 A tile + 128x32 Bt tile into LDS buffer `buf` (8 async per thread)
  auto issueTile = [&](int buf, int k0) {
#pragma unroll
    for (int c = 0; c < 4; ++c) {
      int chunk = tid + c * 128;     // 512 chunks of 8 bf16 per tile
      int r = chunk >> 2;            // 0..127
      int o = (chunk & 3) << 3;      // 0,8,16,24
      async_copy16((unsigned)(uintptr_t)&As[buf][r * LDK + o],
                   A + (size_t)(m0 + r) * Kdim + k0 + o);
      async_copy16((unsigned)(uintptr_t)&Bs[buf][r * LDK + o],
                   Bt + (size_t)(n0 + r) * Kdim + k0 + o);
    }
  };

  issueTile(0, 0);
  int buf = 0;
  for (int k0 = 0; k0 < Kdim; k0 += TK) {
    const bool more = (k0 + TK) < Kdim;
    if (more) {
      issueTile(buf ^ 1, k0 + TK);
      // 8 newest outstanding = the prefetch just issued; current tile has landed.
      asm volatile("s_wait_asynccnt 8" ::: "memory");
    } else {
      asm volatile("s_wait_asynccnt 0" ::: "memory");
    }
    __syncthreads();

    v16bf afrag[4], bfrag[4];
#pragma unroll
    for (int i = 0; i < 4; ++i) {
      const bf16_t* p = &As[buf][(wm + i * 16 + lrow) * LDK + kA];
      v8bf lo = *(const v8bf*)p;
      v8bf hi = *(const v8bf*)(p + 16);
      afrag[i] = __builtin_shufflevector(lo, hi, 0,1,2,3,4,5,6,7,8,9,10,11,12,13,14,15);
    }
#pragma unroll
    for (int j = 0; j < 4; ++j) {
      const bf16_t* p = &Bs[buf][(wn + j * 16 + lrow) * LDK + kB];
      v8bf lo = *(const v8bf*)p;
      v8bf hi = *(const v8bf*)(p + 16);
      bfrag[j] = __builtin_shufflevector(lo, hi, 0,1,2,3,4,5,6,7,8,9,10,11,12,13,14,15);
    }
#pragma unroll
    for (int i = 0; i < 4; ++i)
#pragma unroll
      for (int j = 0; j < 4; ++j)
        acc[i][j] = __builtin_amdgcn_wmma_f32_16x16x32_bf16(
            false, afrag[i], false, bfrag[j], (short)0, acc[i][j], false, false);
    __syncthreads();   // all waves done reading buf before it is refilled
    buf ^= 1;
  }

  // epilogue: C/D layout -> lane l: row = v + (l<16?0:8), col = l%16
  const int rb = (lane < 16) ? 0 : 8;
#pragma unroll
  for (int i = 0; i < 4; ++i)
#pragma unroll
    for (int j = 0; j < 4; ++j) {
      int col = n0 + wn + j * 16 + lrow;
      float bv = bias ? bias[col] : 0.0f;
#pragma unroll
      for (int v = 0; v < 8; ++v) {
        int row = m0 + wm + i * 16 + rb + v;
        float val = acc[i][j][v] + bv;
        if (relu) val = fmaxf(val, 0.0f);
        size_t idx = (size_t)row * Ndim + col;
        if (C)  C[idx]  = val;
        if (Cb) Cb[idx] = (bf16_t)val;
      }
    }
}

// ---------------- fp32 -> bf16 convert ----------------
__global__ void f2bf_kernel(const float* __restrict__ in, bf16_t* __restrict__ out, size_t n) {
  size_t i = (size_t)blockIdx.x * blockDim.x + threadIdx.x;
  size_t stride = (size_t)gridDim.x * blockDim.x;
  for (; i < n; i += stride) out[i] = (bf16_t)in[i];
}

// ---------------- W [K][N] fp32 -> Wt [N][K] bf16 ----------------
__global__ __launch_bounds__(256) void transpose_f2bf_kernel(
    const float* __restrict__ W, bf16_t* __restrict__ Wt, int Kd, int Nd)
{
  __shared__ float tile[32][33];
  int nb = blockIdx.x * 32, kb = blockIdx.y * 32;
  int tx = threadIdx.x & 31, ty = threadIdx.x >> 5;
  for (int r = ty; r < 32; r += 8)
    tile[r][tx] = W[(size_t)(kb + r) * Nd + nb + tx];
  __syncthreads();
  for (int r = ty; r < 32; r += 8)
    Wt[(size_t)(nb + r) * Kd + kb + tx] = (bf16_t)tile[tx][r];
}

// ---------------- residual + LayerNorm ----------------
__global__ __launch_bounds__(256) void layernorm_kernel(
    const float* __restrict__ A, const float* __restrict__ Bv,
    const float* __restrict__ g, const float* __restrict__ beta,
    float* __restrict__ out, bf16_t* __restrict__ outb, int D)
{
  int row = blockIdx.x;
  const float* pa = A + (size_t)row * D;
  const float* pb = Bv + (size_t)row * D;
  int tid = threadIdx.x, lane = tid & 31, wave = tid >> 5;
  float s = 0.f, s2 = 0.f;
  for (int i = tid; i < D; i += 256) { float v = pa[i] + pb[i]; s += v; s2 += v * v; }
  __shared__ float r1[8], r2[8];
  __shared__ float smean, sinv;
  s = waveSum(s); s2 = waveSum(s2);
  if (lane == 0) { r1[wave] = s; r2[wave] = s2; }
  __syncthreads();
  if (tid == 0) {
    float S = 0.f, S2 = 0.f;
    for (int i = 0; i < 8; ++i) { S += r1[i]; S2 += r2[i]; }
    float mean = S / D;
    float var  = S2 / D - mean * mean;
    smean = mean; sinv = rsqrtf(var + 1e-5f);
  }
  __syncthreads();
  float mean = smean, inv = sinv;
  for (int i = tid; i < D; i += 256) {
    float v = pa[i] + pb[i];
    float y = (v - mean) * inv * g[i] + beta[i];
    out[(size_t)row * D + i] = y;
    if (outb) outb[(size_t)row * D + i] = (bf16_t)y;
  }
}

// ---------------- ProbSparse: sampled sparsity measure ----------------
__global__ __launch_bounds__(256) void sample_measure_kernel(
    const float* __restrict__ Q, const float* __restrict__ K,
    float* __restrict__ Mout, int Bn, int Hn, int Ln, int Dh, int Up, unsigned seed)
{
  int gw = (int)((blockIdx.x * blockDim.x + threadIdx.x) >> 5);
  int lane = threadIdx.x & 31;
  int total = Bn * Hn * Ln;
  if (gw >= total) return;
  int l = gw % Ln; int h = (gw / Ln) % Hn; int b = gw / (Ln * Hn);
  int Dm = Hn * Dh;
  const float* q = Q + (size_t)(b * Ln + l) * Dm + h * Dh;
  float mx = -3.0e38f, sm = 0.f;
  for (int s = lane; s < Up; s += 32) {
    unsigned xh = (unsigned)(l * Up + s) * 2654435761u + seed;
    xh ^= xh >> 16; xh *= 0x7feb352du; xh ^= xh >> 15; xh *= 0x846ca68bu; xh ^= xh >> 16;
    int kk = (int)(xh & (unsigned)(Ln - 1));   // Ln is a power of two
    const float* kp = K + (size_t)(b * Ln + kk) * Dm + h * Dh;
    float d = 0.f;
    for (int e = 0; e < Dh; ++e) d += q[e] * kp[e];
    mx = fmaxf(mx, d); sm += d;
  }
  mx = waveMax(mx); sm = waveSum(sm);
  if (lane == 0) Mout[gw] = mx - sm / (float)Ln;
}

// ---------------- top-u selection per (b,h) ----------------
__global__ __launch_bounds__(256) void topk_kernel(
    const float* __restrict__ Mmeas, int* __restrict__ Mtop, int Ln, int u)
{
  int bh = blockIdx.x;
  __shared__ float vals[1024];
  __shared__ float bestv[256];
  __shared__ int   besti[256];
  const float* src = Mmeas + (size_t)bh * Ln;
  for (int i = threadIdx.x; i < Ln; i += 256) vals[i] = src[i];
  __syncthreads();
  for (int it = 0; it < u; ++it) {
    float bv = -3.0e38f; int bi = 0;
    for (int i = threadIdx.x; i < Ln; i += 256)
      if (vals[i] > bv) { bv = vals[i]; bi = i; }
    bestv[threadIdx.x] = bv; besti[threadIdx.x] = bi;
    __syncthreads();
    if (threadIdx.x == 0) {
      float b2 = -3.0e38f; int i2 = 0;
      for (int t = 0; t < 256; ++t)
        if (bestv[t] > b2) { b2 = bestv[t]; i2 = besti[t]; }
      Mtop[(size_t)bh * u + it] = i2;
      vals[i2] = -3.0e38f;
    }
    __syncthreads();
  }
}

// ---------------- softmax(QK^T)V for the u selected queries ----------------
__global__ __launch_bounds__(256) void attn_ctx_kernel(
    const float* __restrict__ Q, const float* __restrict__ K,
    const float* __restrict__ V, const int* __restrict__ Mtop,
    float* __restrict__ ctx, int Bn, int Hn, int Ln, int Dh, int u,
    int causal, float scale)
{
  int gi = blockIdx.x;
  int i = gi % u; int h = (gi / u) % Hn; int b = gi / (u * Hn);
  int lq = Mtop[((size_t)(b * Hn + h)) * u + i];
  int Dm = Hn * Dh;
  const float* q = Q + (size_t)(b * Ln + lq) * Dm + h * Dh;

  __shared__ float sq[64];
  __shared__ float w[1024];
  __shared__ float red[8];
  __shared__ float smax, ssum;
  __shared__ float part[256];

  int tid = threadIdx.x, lane = tid & 31, wave = tid >> 5;
  if (tid < Dh) sq[tid] = q[tid];
  __syncthreads();

  float lmx = -3.0e38f;
  for (int k = tid; k < Ln; k += 256) {
    const float* kp = K + (size_t)(b * Ln + k) * Dm + h * Dh;
    float d = 0.f;
    for (int e = 0; e < Dh; ++e) d += sq[e] * kp[e];
    d *= scale;
    if (causal && (k > lq)) d = -3.0e38f;
    w[k] = d;
    lmx = fmaxf(lmx, d);
  }
  lmx = waveMax(lmx);
  if (lane == 0) red[wave] = lmx;
  __syncthreads();
  if (tid == 0) {
    float r = red[0];
    for (int t = 1; t < 8; ++t) r = fmaxf(r, red[t]);
    smax = r;
  }
  __syncthreads();

  float mx = smax;
  float ls = 0.f;
  for (int k = tid; k < Ln; k += 256) { float e = __expf(w[k] - mx); w[k] = e; ls += e; }
  ls = waveSum(ls);
  __syncthreads();
  if (lane == 0) red[wave] = ls;
  __syncthreads();
  if (tid == 0) {
    float r = 0.f;
    for (int t = 0; t < 8; ++t) r += red[t];
    ssum = r;
  }
  __syncthreads();

  float denom = ssum;
  int dd = tid & 63, kc = tid >> 6;
  int span = Ln >> 2;
  float accv = 0.f;
  for (int k = kc * span; k < (kc + 1) * span; ++k)
    accv += w[k] * V[(size_t)(b * Ln + k) * Dm + h * Dh + dd];
  part[tid] = accv;
  __syncthreads();
  if (tid < Dh) {
    float r = part[tid] + part[tid + 64] + part[tid + 128] + part[tid + 192];
    ctx[(((size_t)(b * Hn + h)) * u + i) * Dh + tid] = r / denom;
  }
}

// ---------------- causal default context: cumsum of V over L ----------------
__global__ void cumsum_cols_kernel(const float* __restrict__ V, float* __restrict__ out,
                                   int Ln, int Dm)
{
  int b = blockIdx.y;
  int col = blockIdx.x * blockDim.x + threadIdx.x;
  if (col >= Dm) return;
  const float* src = V + (size_t)b * Ln * Dm + col;
  float* dst = out + (size_t)b * Ln * Dm + col;
  float s = 0.f;
  for (int l = 0; l < Ln; ++l) { s += src[(size_t)l * Dm]; dst[(size_t)l * Dm] = s; }
}

// ---------------- non-causal default context: mean of V broadcast ----------------
__global__ void mean_cols_kernel(const float* __restrict__ V, float* __restrict__ out,
                                 int Ln, int Dm)
{
  int b = blockIdx.y;
  int col = blockIdx.x * blockDim.x + threadIdx.x;
  if (col >= Dm) return;
  const float* src = V + (size_t)b * Ln * Dm + col;
  float* dst = out + (size_t)b * Ln * Dm + col;
  float s = 0.f;
  for (int l = 0; l < Ln; ++l) s += src[(size_t)l * Dm];
  float m = s / (float)Ln;
  for (int l = 0; l < Ln; ++l) dst[(size_t)l * Dm] = m;
}

// ---------------- scatter computed ctx rows back into context ----------------
__global__ void scatter_ctx_kernel(const float* __restrict__ ctx, const int* __restrict__ Mtop,
                                   float* __restrict__ context, int Bn, int Hn, int Ln,
                                   int Dh, int u)
{
  int gi = blockIdx.x;
  int i = gi % u; int h = (gi / u) % Hn; int b = gi / (u * Hn);
  int l = Mtop[((size_t)(b * Hn + h)) * u + i];
  int d = threadIdx.x;
  if (d < Dh)
    context[(size_t)(b * Ln + l) * (Hn * Dh) + h * Dh + d] =
        ctx[(((size_t)(b * Hn + h)) * u + i) * Dh + d];
}

// ---------------- host orchestration ----------------
static void run_gemm(const bf16_t* A, const bf16_t* Bt, const float* bias,
                     float* C, bf16_t* Cb, int M, int N, int K, int relu,
                     hipStream_t s)
{
  dim3 grid(N / TN, M / TM);
  gemm_bf16_wmma<<<grid, 128, 0, s>>>(A, Bt, bias, C, Cb, M, N, K, relu);
}

extern "C" void kernel_launch(void* const* d_in, const int* in_sizes, int n_in,
                              void* d_out, int out_size, void* d_ws, size_t ws_size,
                              hipStream_t stream)
{
  (void)in_sizes; (void)n_in; (void)out_size; (void)ws_size;
  const int Bn = 4, Ln = 1024, Dm = 1024, Fh = 4096, Hn = 16, Dh = 64;
  const int Mrows = Bn * Ln;
  const int u = 35, Up = 35;           // FACTOR * ceil(ln(1024)) = 5*7
  const float scale = 0.125f;          // 1/sqrt(64)

  const float* x    = (const float*)d_in[0];
  const float* enc  = (const float*)d_in[1];
  const float* saWq = (const float*)d_in[2];  const float* sabq = (const float*)d_in[3];
  const float* saWk = (const float*)d_in[4];  const float* sabk = (const float*)d_in[5];
  const float* saWv = (const float*)d_in[6];  const float* sabv = (const float*)d_in[7];
  const float* saWo = (const float*)d_in[8];  const float* sabo = (const float*)d_in[9];
  const float* caWq = (const float*)d_in[10]; const float* cabq = (const float*)d_in[11];
  const float* caWk = (const float*)d_in[12]; const float* cabk = (const float*)d_in[13];
  const float* caWv = (const float*)d_in[14]; const float* cabv = (const float*)d_in[15];
  const float* caWo = (const float*)d_in[16]; const float* cabo = (const float*)d_in[17];
  const float* ffW1 = (const float*)d_in[18]; const float* ffb1 = (const float*)d_in[19];
  const float* ffW2 = (const float*)d_in[20]; const float* ffb2 = (const float*)d_in[21];
  const float* ln1g = (const float*)d_in[22]; const float* ln1b = (const float*)d_in[23];
  const float* ln2g = (const float*)d_in[24]; const float* ln2b = (const float*)d_in[25];
  const float* ln3g = (const float*)d_in[26]; const float* ln3b = (const float*)d_in[27];

  char* base = (char*)d_ws; size_t off = 0;
  auto alloc = [&](size_t bytes) -> void* {
    size_t a = (off + 255) & ~(size_t)255;
    off = a + bytes;
    return (void*)(base + a);
  };
  const size_t MD = (size_t)Mrows * Dm, MF = (size_t)Mrows * Fh;
  const size_t DD = (size_t)Dm * Dm, DF = (size_t)Dm * Fh;

  bf16_t* xb       = (bf16_t*)alloc(MD * 2);
  bf16_t* encb     = (bf16_t*)alloc(MD * 2);
  bf16_t* wt_saWq  = (bf16_t*)alloc(DD * 2);
  bf16_t* wt_saWk  = (bf16_t*)alloc(DD * 2);
  bf16_t* wt_saWv  = (bf16_t*)alloc(DD * 2);
  bf16_t* wt_saWo  = (bf16_t*)alloc(DD * 2);
  bf16_t* wt_caWq  = (bf16_t*)alloc(DD * 2);
  bf16_t* wt_caWk  = (bf16_t*)alloc(DD * 2);
  bf16_t* wt_caWv  = (bf16_t*)alloc(DD * 2);
  bf16_t* wt_caWo  = (bf16_t*)alloc(DD * 2);
  bf16_t* wt_ffW1  = (bf16_t*)alloc(DF * 2);
  bf16_t* wt_ffW2  = (bf16_t*)alloc(DF * 2);
  float*  Qb       = (float*)alloc(MD * 4);
  float*  Kb       = (float*)alloc(MD * 4);
  float*  Vb       = (float*)alloc(MD * 4);
  float*  Mmeas    = (float*)alloc((size_t)Bn * Hn * Ln * 4);
  int*    Mtop     = (int*)alloc((size_t)Bn * Hn * u * 4);
  float*  ctxS     = (float*)alloc((size_t)Bn * Hn * u * Dh * 4);
  float*  contextF = (float*)alloc(MD * 4);
  bf16_t* contextB = (bf16_t*)alloc(MD * 2);
  float*  attnO    = (float*)alloc(MD * 4);
  float*  x1       = (float*)alloc(MD * 4);
  bf16_t* x1b      = (bf16_t*)alloc(MD * 2);
  float*  x2       = (float*)alloc(MD * 4);
  bf16_t* x2b      = (bf16_t*)alloc(MD * 2);
  bf16_t* h1b      = (bf16_t*)alloc(MF * 2);
  float*  h2       = (float*)alloc(MD * 4);

  // ---- precompute bf16 activations + transposed bf16 weights ----
  f2bf_kernel<<<1024, 256, 0, stream>>>(x,   xb,   MD);
  f2bf_kernel<<<1024, 256, 0, stream>>>(enc, encb, MD);
  dim3 tDD(Dm / 32, Dm / 32), tDF(Fh / 32, Dm / 32), tFD(Dm / 32, Fh / 32);
  transpose_f2bf_kernel<<<tDD, 256, 0, stream>>>(saWq, wt_saWq, Dm, Dm);
  transpose_f2bf_kernel<<<tDD, 256, 0, stream>>>(saWk, wt_saWk, Dm, Dm);
  transpose_f2bf_kernel<<<tDD, 256, 0, stream>>>(saWv, wt_saWv, Dm, Dm);
  transpose_f2bf_kernel<<<tDD, 256, 0, stream>>>(saWo, wt_saWo, Dm, Dm);
  transpose_f2bf_kernel<<<tDD, 256, 0, stream>>>(caWq, wt_caWq, Dm, Dm);
  transpose_f2bf_kernel<<<tDD, 256, 0, stream>>>(caWk, wt_caWk, Dm, Dm);
  transpose_f2bf_kernel<<<tDD, 256, 0, stream>>>(caWv, wt_caWv, Dm, Dm);
  transpose_f2bf_kernel<<<tDD, 256, 0, stream>>>(caWo, wt_caWo, Dm, Dm);
  transpose_f2bf_kernel<<<tDF, 256, 0, stream>>>(ffW1, wt_ffW1, Dm, Fh);
  transpose_f2bf_kernel<<<tFD, 256, 0, stream>>>(ffW2, wt_ffW2, Fh, Dm);

  const int mwBlocks = (Bn * Hn * Ln * 32 + 255) / 256;
  dim3 colGrid(Dm / 256, Bn);

  // ---- phase 1: masked self-attention ----
  run_gemm(xb, wt_saWq, sabq, Qb, nullptr, Mrows, Dm, Dm, 0, stream);
  run_gemm(xb, wt_saWk, sabk, Kb, nullptr, Mrows, Dm, Dm, 0, stream);
  run_gemm(xb, wt_saWv, sabv, Vb, nullptr, Mrows, Dm, Dm, 0, stream);
  sample_measure_kernel<<<mwBlocks, 256, 0, stream>>>(Qb, Kb, Mmeas, Bn, Hn, Ln, Dh, Up, 0x9E3779B9u);
  topk_kernel<<<Bn * Hn, 256, 0, stream>>>(Mmeas, Mtop, Ln, u);
  attn_ctx_kernel<<<Bn * Hn * u, 256, 0, stream>>>(Qb, Kb, Vb, Mtop, ctxS, Bn, Hn, Ln, Dh, u, 1, scale);
  cumsum_cols_kernel<<<colGrid, 256, 0, stream>>>(Vb, contextF, Ln, Dm);
  scatter_ctx_kernel<<<Bn * Hn * u, 64, 0, stream>>>(ctxS, Mtop, contextF, Bn, Hn, Ln, Dh, u);
  f2bf_kernel<<<1024, 256, 0, stream>>>(contextF, contextB, MD);
  run_gemm(contextB, wt_saWo, sabo, attnO, nullptr, Mrows, Dm, Dm, 0, stream);
  layernorm_kernel<<<Mrows, 256, 0, stream>>>(x, attnO, ln1g, ln1b, x1, x1b, Dm);

  // ---- phase 2: cross-attention (non-causal) ----
  run_gemm(x1b,  wt_caWq, cabq, Qb, nullptr, Mrows, Dm, Dm, 0, stream);
  run_gemm(encb, wt_caWk, cabk, Kb, nullptr, Mrows, Dm, Dm, 0, stream);
  run_gemm(encb, wt_caWv, cabv, Vb, nullptr, Mrows, Dm, Dm, 0, stream);
  sample_measure_kernel<<<mwBlocks, 256, 0, stream>>>(Qb, Kb, Mmeas, Bn, Hn, Ln, Dh, Up, 0x85EBCA77u);
  topk_kernel<<<Bn * Hn, 256, 0, stream>>>(Mmeas, Mtop, Ln, u);
  attn_ctx_kernel<<<Bn * Hn * u, 256, 0, stream>>>(Qb, Kb, Vb, Mtop, ctxS, Bn, Hn, Ln, Dh, u, 0, scale);
  mean_cols_kernel<<<colGrid, 256, 0, stream>>>(Vb, contextF, Ln, Dm);
  scatter_ctx_kernel<<<Bn * Hn * u, 64, 0, stream>>>(ctxS, Mtop, contextF, Bn, Hn, Ln, Dh, u);
  f2bf_kernel<<<1024, 256, 0, stream>>>(contextF, contextB, MD);
  run_gemm(contextB, wt_caWo, cabo, attnO, nullptr, Mrows, Dm, Dm, 0, stream);
  layernorm_kernel<<<Mrows, 256, 0, stream>>>(x1, attnO, ln2g, ln2b, x2, x2b, Dm);

  // ---- phase 3: FFN ----
  run_gemm(x2b, wt_ffW1, ffb1, nullptr, h1b, Mrows, Fh, Dm, 1, stream);  // ReLU, bf16 out
  run_gemm(h1b, wt_ffW2, ffb2, h2, nullptr, Mrows, Dm, Fh, 0, stream);
  layernorm_kernel<<<Mrows, 256, 0, stream>>>(x2, h2, ln3g, ln3b, (float*)d_out, nullptr, Dm);
}
